// NGDCSDE_40888088658174
// MI455X (gfx1250) — compile-verified, matrix operand order checked by reading, not run on hardware
//
#include <hip/hip_runtime.h>
#include <math.h>

// ---------------------------------------------------------------------------
// NGDCSDE forward for MI455X (gfx1250, wave32, WMMA f32_16x16x32_f16).
// GEMM tiling: block = 128 threads (4 waves), tile = 64(M) x 32(N).
// Each wave owns 16 rows and both 16-col halves (2 accumulators, A reused).
// X tile staged k-major, W tile staged TRANSPOSED (n-major) in LDS so both
// A and B fragments are contiguous 16B ds_load_b128 reads.
// All shapes: M % 64 == 0, Ncol % 32 == 0, K % 32 == 0 -> no remainders.
// ---------------------------------------------------------------------------

typedef __attribute__((ext_vector_type(16))) _Float16 v16h;
typedef __attribute__((ext_vector_type(8)))  _Float16 v8h;
typedef __attribute__((ext_vector_type(8)))  float    v8f;

#define B_   8
#define T_   12
#define N_   512
#define DIN_ 32
#define DR_  96
#define DH_  128
#define S_   8
#define BN_  (B_ * N_)    // 4096
#define TBN_ (T_ * BN_)   // 49152
#define SBN_ (S_ * BN_)   // 32768

#define XS_LD 48   // LDS row stride (halves) for 64x32 X tile (96B, 16B-aligned)
#define WS_LD 48   // LDS row stride (halves) for transposed 32x32 W tile

__device__ __forceinline__ float sigf(float x) {
  return __builtin_amdgcn_rcpf(1.0f + __expf(-x));
}
__device__ __forceinline__ float tanh_fast(float x) {
  x = fminf(15.0f, fmaxf(-15.0f, x));
  float e = __expf(2.0f * x);
  return (e - 1.0f) * __builtin_amdgcn_rcpf(e + 1.0f);
}
__device__ __forceinline__ float act_apply(float v, int act) {
  if (act == 1) return tanh_fast(v);
  if (act == 2) return sigf(v);
  return v;
}

// Cooperative f32->f16 tile loaders ------------------------------------------
// X tile: 64 rows x 32 k, thread tid loads 16 consecutive floats of one row.
__device__ __forceinline__ void load_xtile(_Float16* Xs, const float* Xrow0,
                                           int ldx, int k0, int tid, int Kd) {
  const int r  = tid >> 1;
  const int cb = (tid & 1) * 16;
  const float* xp = Xrow0 + (long)r * ldx + k0 + cb;
  float4 a = ((const float4*)xp)[0];
  float4 b = ((const float4*)xp)[1];
  float4 c = ((const float4*)xp)[2];
  float4 d = ((const float4*)xp)[3];
  if (k0 + 32 < Kd) __builtin_prefetch(xp + 32, 0, 0);
  _Float16* s = Xs + r * XS_LD + cb;
  s[0]=(_Float16)a.x; s[1]=(_Float16)a.y; s[2]=(_Float16)a.z; s[3]=(_Float16)a.w;
  s[4]=(_Float16)b.x; s[5]=(_Float16)b.y; s[6]=(_Float16)b.z; s[7]=(_Float16)b.w;
  s[8]=(_Float16)c.x; s[9]=(_Float16)c.y; s[10]=(_Float16)c.z; s[11]=(_Float16)c.w;
  s[12]=(_Float16)d.x; s[13]=(_Float16)d.y; s[14]=(_Float16)d.z; s[15]=(_Float16)d.w;
}
// W tile: 32 k-rows x 32 n cols, stored TRANSPOSED: Wt[n][k], n-major.
// Thread tid loads W[k0+kr][n0+cb .. +8) coalesced, scatters into column kr.
__device__ __forceinline__ void load_wtile_t(_Float16* Wt, const float* Wbase,
                                             int ldw, int k0, int n0, int tid, int Kd) {
  const int kr = tid >> 2;
  const int cb = (tid & 3) * 8;
  const float* wp = Wbase + (long)(k0 + kr) * ldw + n0 + cb;
  float4 a = ((const float4*)wp)[0];
  float4 b = ((const float4*)wp)[1];
  if (k0 + 32 < Kd) __builtin_prefetch(wp + (long)32 * ldw, 0, 0);
  _Float16* s = Wt + cb * WS_LD + kr;
  s[0 * WS_LD] = (_Float16)a.x; s[1 * WS_LD] = (_Float16)a.y;
  s[2 * WS_LD] = (_Float16)a.z; s[3 * WS_LD] = (_Float16)a.w;
  s[4 * WS_LD] = (_Float16)b.x; s[5 * WS_LD] = (_Float16)b.y;
  s[6 * WS_LD] = (_Float16)b.z; s[7 * WS_LD] = (_Float16)b.w;
}
// Fragment from one LDS row (ISA 7.12.2 16-bit layout): halves 0..7 = K kb+0..7,
// halves 8..15 = K kb+16..23. Two 16B-aligned v8h (ds_load_b128) reads.
__device__ __forceinline__ v16h frag_from(const _Float16* rowp, int kb) {
  const v8h lo = *(const v8h*)(rowp + kb);
  const v8h hi = *(const v8h*)(rowp + kb + 16);
  v16h r;
#pragma unroll
  for (int i = 0; i < 8; ++i) { r[i] = lo[i]; r[i + 8] = hi[i]; }
  return r;
}

// ---------------------------------------------------------------------------
// Generic GEMM: C[M,Nc] = act(X[M,Kd] @ W[Kd,Nc] + bias), strided X/C.
// grid (Ncol/32, M/64), block 128.
// ---------------------------------------------------------------------------
__global__ __launch_bounds__(128) void k_gemm(
    const float* __restrict__ X, const float* __restrict__ W,
    const float* __restrict__ bias, float* __restrict__ C,
    int M, int Kd, int Ncol, int ldx, int ldc, int act)
{
  __shared__ __align__(16) _Float16 Xs[64 * XS_LD];
  __shared__ __align__(16) _Float16 Wt[32 * WS_LD];
  const int tid = threadIdx.x;
  const unsigned lane = tid & 31u;
  const int wv = tid >> 5;
  const int n0 = blockIdx.x * 32;
  const int mblk = blockIdx.y * 64;
  const int rr = (int)(lane & 15u);
  const int kb = (int)(lane >> 4) * 8;
  const float* Xrow0 = X + (long)mblk * ldx;

  v8f acc0 = {}, acc1 = {};
  for (int k0 = 0; k0 < Kd; k0 += 32) {
    load_xtile(Xs, Xrow0, ldx, k0, tid, Kd);
    load_wtile_t(Wt, W, Ncol, k0, n0, tid, Kd);
    __syncthreads();
    v16h av  = frag_from(Xs + (wv * 16 + rr) * XS_LD, kb);
    v16h bv0 = frag_from(Wt + rr * WS_LD, kb);
    v16h bv1 = frag_from(Wt + (16 + rr) * WS_LD, kb);
    acc0 = __builtin_amdgcn_wmma_f32_16x16x32_f16(false, av, false, bv0,
                                                  (short)0, acc0, false, false);
    acc1 = __builtin_amdgcn_wmma_f32_16x16x32_f16(false, av, false, bv1,
                                                  (short)0, acc1, false, false);
    __syncthreads();
  }
  const int rb = (int)(lane >> 4) * 8;
  const float b0 = bias ? bias[n0 + rr] : 0.0f;
  const float b1 = bias ? bias[n0 + 16 + rr] : 0.0f;
  float* Cp = C + (long)(mblk + wv * 16) * ldc + n0 + rr;
#pragma unroll
  for (int r = 0; r < 8; ++r) {
    Cp[(long)(rb + r) * ldc]      = act_apply(acc0[r] + b0, act);
    Cp[(long)(rb + r) * ldc + 16] = act_apply(acc1[r] + b1, act);
  }
}

// ---------------------------------------------------------------------------
// Batched diffusion hop: C[b] = A[b] @ X[b], A: [B,512,512], X/C row stride ld.
// grid (F/32, 512/64, B), block 128.
// ---------------------------------------------------------------------------
__global__ __launch_bounds__(128) void k_bmm(
    const float* __restrict__ A, const float* __restrict__ X,
    float* __restrict__ C, int F, int ld)
{
  __shared__ __align__(16) _Float16 Xs[64 * XS_LD];
  __shared__ __align__(16) _Float16 Wt[32 * WS_LD];
  const int b = blockIdx.z;
  const int tid = threadIdx.x;
  const unsigned lane = tid & 31u;
  const int wv = tid >> 5;
  const int n0 = blockIdx.x * 32;
  const int mblk = blockIdx.y * 64;
  const int rr = (int)(lane & 15u);
  const int kb = (int)(lane >> 4) * 8;
  const float* Arow0 = A + (long)b * N_ * N_ + (long)mblk * N_;
  const float* Xb    = X + (long)b * N_ * ld;

  v8f acc0 = {}, acc1 = {};
  for (int k0 = 0; k0 < N_; k0 += 32) {
    load_xtile(Xs, Arow0, N_, k0, tid, N_);
    load_wtile_t(Wt, Xb, ld, k0, n0, tid, N_);
    __syncthreads();
    v16h av  = frag_from(Xs + (wv * 16 + rr) * XS_LD, kb);
    v16h bv0 = frag_from(Wt + rr * WS_LD, kb);
    v16h bv1 = frag_from(Wt + (16 + rr) * WS_LD, kb);
    acc0 = __builtin_amdgcn_wmma_f32_16x16x32_f16(false, av, false, bv0,
                                                  (short)0, acc0, false, false);
    acc1 = __builtin_amdgcn_wmma_f32_16x16x32_f16(false, av, false, bv1,
                                                  (short)0, acc1, false, false);
    __syncthreads();
  }
  const int rb = (int)(lane >> 4) * 8;
  float* Cp = C + (long)b * N_ * ld + (long)(mblk + wv * 16) * ld + n0 + rr;
#pragma unroll
  for (int r = 0; r < 8; ++r) {
    Cp[(long)(rb + r) * ld]      = acc0[r];
    Cp[(long)(rb + r) * ld + 16] = acc1[r];
  }
}

// ---------------------- elementwise helper kernels -------------------------
__global__ void k_ew_add(const float* a, const float* b, float* c, long n) {
  long i = (long)blockIdx.x * blockDim.x + threadIdx.x;
  if (i < n) c[i] = a[i] + b[i];
}
__global__ void k_ew_mul(const float* a, const float* b, float* c, long n) {
  long i = (long)blockIdx.x * blockDim.x + threadIdx.x;
  if (i < n) c[i] = a[i] * b[i];
}
__global__ void k_fill0(float* p, long n) {
  long i = (long)blockIdx.x * blockDim.x + threadIdx.x;
  if (i < n) p[i] = 0.0f;
}
__global__ void k_bcast_vec(float* dst, const float* v, long n, int f) {
  long i = (long)blockIdx.x * blockDim.x + threadIdx.x;
  if (i < n) dst[i] = v[i % f];
}
// dst[m, doff:doff+F) = src[m, 0:F)
__global__ void k_copy_cols(float* dst, int ldd, int doff,
                            const float* src, int lds, long M, int F) {
  long i = (long)blockIdx.x * blockDim.x + threadIdx.x;
  if (i >= M * (long)F) return;
  long m = i / F; int f = (int)(i % F);
  dst[m * ldd + doff + f] = src[m * lds + f];
}
// dst[bn, 0:32) = values[b,t,n,:] * masks[b,t,n,:]
__global__ void k_copy_x(float* dst, int ldd, const float* values,
                         const float* masks, int t) {
  long i = (long)blockIdx.x * blockDim.x + threadIdx.x;
  if (i >= (long)BN_ * DIN_) return;
  long bn = i / DIN_; int d = (int)(i % DIN_);
  long b = bn >> 9, n = bn & 511;
  long s = (((b * T_) + t) * N_ + n) * DIN_ + d;
  dst[bn * ldd + d] = values[s] * masks[s];
}
__global__ void k_obs(const float* __restrict__ masks, float* obs, int t) {
  long i = (long)blockIdx.x * blockDim.x + threadIdx.x;
  if (i >= BN_) return;
  long b = i >> 9, n = i & 511;
  const float* mp = masks + (((b * T_) + t) * N_ + n) * DIN_;
  float s = 0.0f;
  for (int d = 0; d < DIN_; ++d) s += fabsf(mp[d]);
  obs[i] = (s > 1e-4f) ? 1.0f : 0.0f;
}
// rh = ru[:, :96] * h  (ru already sigmoided, ld 192)
__global__ void k_rh(const float* ru, const float* h, float* rh) {
  long i = (long)blockIdx.x * blockDim.x + threadIdx.x;
  if (i >= (long)BN_ * DR_) return;
  long row = i / DR_; int c = (int)(i % DR_);
  rh[i] = ru[row * 192 + c] * h[i];
}
// u = ru[:,96:192]; hnew = u*h1 + (1-u)*cc; state = h1*(1-obs) + hnew*obs
__global__ void k_blend(const float* ru, const float* cc, const float* h1,
                        const float* obs, float* hs) {
  long i = (long)blockIdx.x * blockDim.x + threadIdx.x;
  if (i >= (long)BN_ * DR_) return;
  long row = i / DR_; int c = (int)(i % DR_);
  float u = ru[row * 192 + 96 + c];
  float h1v = h1[i];
  float hnew = u * h1v + (1.0f - u) * cc[i];
  float o = obs[row];
  hs[i] = h1v * (1.0f - o) + hnew * o;
}
__global__ void k_gru_enc(const float* gi, const float* gh, float* h, float* zpre_t) {
  long i = (long)blockIdx.x * blockDim.x + threadIdx.x;
  if (i >= (long)BN_ * DR_) return;
  long row = i / DR_; int c = (int)(i % DR_);
  const float* gip = gi + row * 288;
  const float* ghp = gh + row * 288;
  float r  = sigf(gip[c] + ghp[c]);
  float zg = sigf(gip[96 + c] + ghp[96 + c]);
  float nn = tanh_fast(gip[192 + c] + r * ghp[192 + c]);
  float hv = (1.0f - zg) * nn + zg * h[i];
  h[i] = hv;
  zpre_t[i] = hv;
}
// inp[s*BN+bn, 0:96)=zp[bn,:], [96:192)=Z[s*BN+bn,:]
__global__ void k_sde_inp(const float* zp, const float* Z, float* inp) {
  long i = (long)blockIdx.x * blockDim.x + threadIdx.x;
  if (i >= (long)SBN_ * 192) return;
  long row = i / 192; int c = (int)(i % 192);
  long bn = row % BN_;
  inp[i] = (c < 96) ? zp[bn * DR_ + c] : Z[row * DR_ + (c - 96)];
}
__global__ void k_rownorm(const float* Z, float* Zn) {
  long r = (long)blockIdx.x * blockDim.x + threadIdx.x;
  if (r >= SBN_) return;
  const float* zp = Z + r * DR_;
  float m = zp[0];
  for (int c = 1; c < DR_; ++c) m = fmaxf(m, zp[c]);
  float rm = __builtin_amdgcn_rcpf(m);
  float* op = Zn + r * DR_;
  for (int c = 0; c < DR_; ++c) op[c] = zp[c] * rm;
}
__global__ void k_mean_s(const float* Zn, float* zbar) {
  long i = (long)blockIdx.x * blockDim.x + threadIdx.x;
  if (i >= (long)BN_ * DR_) return;
  long bn = i / DR_; int c = (int)(i % DR_);
  float s = 0.0f;
  for (int ss = 0; ss < S_; ++ss) s += Zn[((long)ss * BN_ + bn) * DR_ + c];
  zbar[i] = s * (1.0f / S_);
}
// out[b, t-1, n, :] = src[bn, :]
__global__ void k_scatter_out(const float* src, float* out, int t) {
  long i = (long)blockIdx.x * blockDim.x + threadIdx.x;
  if (i >= (long)BN_ * DIN_) return;
  long bn = i / DIN_; int c = (int)(i % DIN_);
  long b = bn >> 9, n = bn & 511;
  out[(((b * (T_ - 1)) + (t - 1)) * N_ + n) * DIN_ + c] = src[i];
}

// ---------------------------------------------------------------------------
extern "C" void kernel_launch(void* const* d_in, const int* in_sizes, int n_in,
                              void* d_out, int out_size, void* d_ws, size_t ws_size,
                              hipStream_t stream) {
  auto F = [&](int i) { return (const float*)d_in[i]; };
  const float *A, *masks, *values;
  const float *gVruW,*gVrub,*gVcW,*gVcb,*gGruW,*gGrub,*gGcW,*gGcb;
  const float *oVW0,*oVb0,*oVW1,*oVb1,*oGW0,*oGb0,*oGW1,*oGb1;
  const float *zoW0,*zob0,*zoW1,*zob1,*zoW2,*zob2;
  const float *h0v,*z0v,*z0sv;
  const float *encWi,*encWh,*encbi,*encbh;
  const float *pW0,*pb0,*pW1,*pb1,*dW0,*db0,*dW1,*db1;
  const float *outW,*outb;
  (void)n_in; (void)out_size; (void)ws_size;

  if (in_sizes[0] == B_ * N_ * N_) {
    // JAX pytree flatten order (dict keys sorted at every level)
    A = F(0); masks = F(1);
    dW0 = F(2);  db0 = F(3);  dW1 = F(4);  db1 = F(5);
    encWh = F(6); encWi = F(7); encbh = F(8); encbi = F(9);
    gGcW = F(10); gGcb = F(11); gGruW = F(12); gGrub = F(13);
    gVcW = F(14); gVcb = F(15); gVruW = F(16); gVrub = F(17);
    h0v = F(18);
    oGW0 = F(19); oGb0 = F(20); oGW1 = F(21); oGb1 = F(22);
    oVW0 = F(23); oVb0 = F(24); oVW1 = F(25); oVb1 = F(26);
    outW = F(27); outb = F(28);
    pW0 = F(29); pb0 = F(30); pW1 = F(31); pb1 = F(32);
    z0v = F(33); z0sv = F(34);
    zoW0 = F(35); zob0 = F(36); zoW1 = F(37); zob1 = F(38); zoW2 = F(39); zob2 = F(40);
    values = F(41);
  } else {
    // dict insertion order from setup_inputs()
    values = F(0); masks = F(1); A = F(2);
    gVruW = F(3); gVrub = F(4); gVcW = F(5); gVcb = F(6);
    gGruW = F(7); gGrub = F(8); gGcW = F(9); gGcb = F(10);
    oVW0 = F(11); oVb0 = F(12); oVW1 = F(13); oVb1 = F(14);
    oGW0 = F(15); oGb0 = F(16); oGW1 = F(17); oGb1 = F(18);
    zoW0 = F(19); zob0 = F(20); zoW1 = F(21); zob1 = F(22); zoW2 = F(23); zob2 = F(24);
    h0v = F(25); z0v = F(26);
    encWi = F(27); encWh = F(28); encbi = F(29); encbh = F(30);
    pW0 = F(31); pb0 = F(32); pW1 = F(33); pb1 = F(34);
    dW0 = F(35); db0 = F(36); dW1 = F(37); db1 = F(38);
    z0sv = F(39);
    outW = F(40); outb = F(41);
  }
  float* out = (float*)d_out;

  // --- workspace carve (deterministic; ~203 MB of f32) ---
  float* ws = (float*)d_ws;
  long off = 0;
  auto carve = [&](long n) { float* p = ws + off; off += n; return p; };
  float* HV_PRE = carve((long)TBN_ * DR_);
  float* ZV_PRE = carve((long)TBN_ * DR_);
  float* HG     = carve((long)TBN_ * DR_);
  float* ZPRE   = carve((long)TBN_ * DR_);
  float* HV     = carve((long)BN_ * DR_);
  float* ZV     = carve((long)BN_ * DR_);
  float* OBS    = carve(BN_);
  float* HOPS   = carve((long)BN_ * 512);
  float* TMP1   = carve((long)SBN_ * 192);
  float* TMP2   = carve((long)SBN_ * 128);
  float* TMP3   = carve((long)SBN_ * 96);
  float* TMP4   = carve((long)SBN_ * 128);
  float* RU     = carve((long)BN_ * 192);
  float* RH     = carve((long)BN_ * DR_);
  float* CC     = carve((long)BN_ * DR_);
  float* HENC   = carve((long)BN_ * DR_);
  float* GI     = carve((long)BN_ * 288);
  float* GH     = carve((long)BN_ * 288);
  float* ZST    = carve((long)SBN_ * DR_);
  float* ZNRM   = carve((long)SBN_ * DR_);
  float* ZBAR   = carve((long)BN_ * DR_);
  float* OUTT   = carve((long)BN_ * DIN_);

  auto EWB = [](long n) { return dim3((unsigned)((n + 255) / 256)); };
  auto gemm = [&](const float* X, const float* W, const float* bias, float* C,
                  int M, int Kd, int Ncol, int ldx, int ldc, int act) {
    k_gemm<<<dim3(Ncol / 32, M / 64), 128, 0, stream>>>(X, W, bias, C, M, Kd,
                                                        Ncol, ldx, ldc, act);
  };
  auto bmm = [&](const float* Xs, float* Cd, int Fdim, int ld) {
    k_bmm<<<dim3(Fdim / 32, N_ / 64, B_), 128, 0, stream>>>(A, Xs, Cd, Fdim, ld);
  };

  // init states
  k_bcast_vec<<<EWB((long)BN_ * DR_), 256, 0, stream>>>(HV, h0v, (long)BN_ * DR_, DR_);
  k_bcast_vec<<<EWB((long)BN_ * DR_), 256, 0, stream>>>(ZV, z0v, (long)BN_ * DR_, DR_);

  // ================= stage 1: ODE + DCGRU scan over T =================
  for (int t = 0; t < T_; ++t) {
    float* HVp = HV_PRE + (long)t * BN_ * DR_;
    float* ZVp = ZV_PRE + (long)t * BN_ * DR_;

    // --- dcode(HV, odeV) -> TMP3 ; hv1 = HV + TMP3 -> HVp
    k_copy_cols<<<EWB((long)BN_ * DR_), 256, 0, stream>>>(HOPS, 384, 0, HV, DR_, BN_, DR_);
    for (int h = 1; h <= 3; ++h) bmm(HOPS + (h - 1) * DR_, HOPS + h * DR_, DR_, 384);
    gemm(HOPS, oVW0, oVb0, TMP2, BN_, 384, DH_, 384, DH_, 1);
    k_copy_cols<<<EWB((long)BN_ * DH_), 256, 0, stream>>>(HOPS, 512, 0, TMP2, DH_, BN_, DH_);
    for (int h = 1; h <= 3; ++h) bmm(HOPS + (h - 1) * DH_, HOPS + h * DH_, DH_, 512);
    gemm(HOPS, oVW1, oVb1, TMP3, BN_, 512, DR_, 512, DR_, 0);
    k_ew_add<<<EWB((long)BN_ * DR_), 256, 0, stream>>>(HV, TMP3, HVp, (long)BN_ * DR_);

    // --- dcode(ZV, odeG) -> zv1 -> ZVp
    k_copy_cols<<<EWB((long)BN_ * DR_), 256, 0, stream>>>(HOPS, 384, 0, ZV, DR_, BN_, DR_);
    for (int h = 1; h <= 3; ++h) bmm(HOPS + (h - 1) * DR_, HOPS + h * DR_, DR_, 384);
    gemm(HOPS, oGW0, oGb0, TMP2, BN_, 384, DH_, 384, DH_, 1);
    k_copy_cols<<<EWB((long)BN_ * DH_), 256, 0, stream>>>(HOPS, 512, 0, TMP2, DH_, BN_, DH_);
    for (int h = 1; h <= 3; ++h) bmm(HOPS + (h - 1) * DH_, HOPS + h * DH_, DH_, 512);
    gemm(HOPS, oGW1, oGb1, TMP3, BN_, 512, DR_, 512, DR_, 0);
    k_ew_add<<<EWB((long)BN_ * DR_), 256, 0, stream>>>(ZV, TMP3, ZVp, (long)BN_ * DR_);

    k_obs<<<EWB(BN_), 256, 0, stream>>>(masks, OBS, t);

    // --- DCGRU V: updates carry HV using x_t and hv1
    k_copy_x<<<EWB((long)BN_ * DIN_), 256, 0, stream>>>(HOPS, 512, values, masks, t);
    k_copy_cols<<<EWB((long)BN_ * DR_), 256, 0, stream>>>(HOPS, 512, DIN_, HVp, DR_, BN_, DR_);
    for (int h = 1; h <= 3; ++h) bmm(HOPS + (h - 1) * 128, HOPS + h * 128, 128, 512);
    gemm(HOPS, gVruW, gVrub, RU, BN_, 512, 192, 512, 192, 2);   // sigmoid(ru)
    k_rh<<<EWB((long)BN_ * DR_), 256, 0, stream>>>(RU, HVp, RH);
    k_copy_x<<<EWB((long)BN_ * DIN_), 256, 0, stream>>>(HOPS, 512, values, masks, t);
    k_copy_cols<<<EWB((long)BN_ * DR_), 256, 0, stream>>>(HOPS, 512, DIN_, RH, DR_, BN_, DR_);
    for (int h = 1; h <= 3; ++h) bmm(HOPS + (h - 1) * 128, HOPS + h * 128, 128, 512);
    gemm(HOPS, gVcW, gVcb, CC, BN_, 512, DR_, 512, DR_, 1);     // tanh(c)
    k_blend<<<EWB((long)BN_ * DR_), 256, 0, stream>>>(RU, CC, HVp, OBS, HV);

    // --- DCGRU G: updates carry ZV
    k_copy_x<<<EWB((long)BN_ * DIN_), 256, 0, stream>>>(HOPS, 512, values, masks, t);
    k_copy_cols<<<EWB((long)BN_ * DR_), 256, 0, stream>>>(HOPS, 512, DIN_, ZVp, DR_, BN_, DR_);
    for (int h = 1; h <= 3; ++h) bmm(HOPS + (h - 1) * 128, HOPS + h * 128, 128, 512);
    gemm(HOPS, gGruW, gGrub, RU, BN_, 512, 192, 512, 192, 2);
    k_rh<<<EWB((long)BN_ * DR_), 256, 0, stream>>>(RU, ZVp, RH);
    k_copy_x<<<EWB((long)BN_ * DIN_), 256, 0, stream>>>(HOPS, 512, values, masks, t);
    k_copy_cols<<<EWB((long)BN_ * DR_), 256, 0, stream>>>(HOPS, 512, DIN_, RH, DR_, BN_, DR_);
    for (int h = 1; h <= 3; ++h) bmm(HOPS + (h - 1) * 128, HOPS + h * 128, 128, 512);
    gemm(HOPS, gGcW, gGcb, CC, BN_, 512, DR_, 512, DR_, 1);
    k_blend<<<EWB((long)BN_ * DR_), 256, 0, stream>>>(RU, CC, ZVp, OBS, ZV);
  }

  // ================= stage 2: zout gate, Hg = hv_pre * gate ============
  gemm(ZV_PRE, zoW0, zob0, TMP1, TBN_, DR_, DR_, DR_, DR_, 1);  // tanh
  gemm(TMP1, zoW1, zob1, ZPRE, TBN_, DR_, DR_, DR_, DR_, 1);    // tanh (ZPRE as scratch)
  gemm(ZPRE, zoW2, zob2, TMP1, TBN_, DR_, DR_, DR_, DR_, 2);    // sigmoid
  k_ew_mul<<<EWB((long)TBN_ * DR_), 256, 0, stream>>>(HV_PRE, TMP1, HG, (long)TBN_ * DR_);

  // ================= stage 3: encoder GRU over T =======================
  k_fill0<<<EWB((long)BN_ * DR_), 256, 0, stream>>>(HENC, (long)BN_ * DR_);
  for (int t = 0; t < T_; ++t) {
    gemm(HG + (long)t * BN_ * DR_, encWi, encbi, GI, BN_, DR_, 288, DR_, 288, 0);
    gemm(HENC, encWh, encbh, GH, BN_, DR_, 288, DR_, 288, 0);
    k_gru_enc<<<EWB((long)BN_ * DR_), 256, 0, stream>>>(GI, GH, HENC,
                                                        ZPRE + (long)t * BN_ * DR_);
  }

  // ================= stage 4+5: latent SDE + fused output ==============
  // mean_S(concat(Hg, Zs) @ Wout) == concat(Hg, mean_S Zs) @ Wout
  k_bcast_vec<<<EWB((long)SBN_ * DR_), 256, 0, stream>>>(ZST, z0sv, (long)SBN_ * DR_, DR_);
  for (int t = 0; t < T_; ++t) {
    k_sde_inp<<<EWB((long)SBN_ * 192), 256, 0, stream>>>(ZPRE + (long)t * BN_ * DR_, ZST, TMP1);
    gemm(TMP1, pW0, pb0, TMP2, SBN_, 192, DH_, 192, DH_, 1);    // tanh
    gemm(TMP2, pW1, pb1, TMP3, SBN_, DH_, DR_, DH_, DR_, 0);    // prior
    gemm(TMP3, dW0, db0, TMP4, SBN_, DR_, DH_, DR_, DH_, 1);    // tanh
    gemm(TMP4, dW1, db1, TMP1, SBN_, DH_, DR_, DH_, DR_, 0);    // drift
    k_ew_add<<<EWB((long)SBN_ * DR_), 256, 0, stream>>>(TMP3, TMP1, ZST, (long)SBN_ * DR_);
    k_rownorm<<<EWB(SBN_), 256, 0, stream>>>(ZST, ZNRM);
    if (t >= 1) {
      k_mean_s<<<EWB((long)BN_ * DR_), 256, 0, stream>>>(ZNRM, ZBAR);
      k_copy_cols<<<EWB((long)BN_ * DR_), 256, 0, stream>>>(TMP2, 192, 0,
          HG + (long)t * BN_ * DR_, DR_, BN_, DR_);
      k_copy_cols<<<EWB((long)BN_ * DR_), 256, 0, stream>>>(TMP2, 192, 96, ZBAR, DR_, BN_, DR_);
      gemm(TMP2, outW, outb, OUTT, BN_, 192, DIN_, 192, DIN_, 0);
      k_scatter_out<<<EWB((long)BN_ * DIN_), 256, 0, stream>>>(OUTT, out, t);
    }
  }
}
